// Bahdanau_85040352461299
// MI455X (gfx1250) — matrix operand order repaired and checked by reading, not compiled, not run on hardware
//
#include <hip/hip_runtime.h>
#include <hip/hip_bf16.h>

// ---------------------------------------------------------------------------
// CDNA5 (gfx1250) wave32 WMMA implementation of a Bahdanau-attention GRU
// decoder.  All GEMMs use v_wmma_f32_16x16x32_bf16 (bf16 in, f32 accum).
// Weights are stored TRANSPOSED (N x K) in bf16 so each lane's B fragment is
// contiguous -> global_load_b128.  The step-invariant WaS / encoder tensors
// are bf16 (2 x 32 MB) and stay resident in the 192 MB L2 for all 256 steps.
// ---------------------------------------------------------------------------

typedef __attribute__((ext_vector_type(16))) __bf16        v16bf;
typedef __attribute__((ext_vector_type(8)))  float         v8f;
typedef __attribute__((ext_vector_type(8)))  unsigned int  v8u;
typedef __attribute__((ext_vector_type(4)))  unsigned int  u32x4;

#define B_SZ    32
#define T_ENC   1024
#define E_DIM   512
#define T_DEC   256
#define D_DIM   256
#define UNITS_N 256
#define OUT_D_N 512
#define GRU3    (3 * UNITS_N)             // 768
#define XIN     (E_DIM + OUT_D_N)         // 1024 (GRU input  [c, out_prev])
#define CAT1    (D_DIM + UNITS_N + E_DIM) // 1024 (dense1 in  [x_t, h, c])

static __device__ __forceinline__ unsigned short f32_to_bf16(float f) {
    unsigned int u = __builtin_bit_cast(unsigned int, f);
    unsigned int lsb = (u >> 16) & 1u;
    u += 0x7fffu + lsb;                     // round-to-nearest-even
    return (unsigned short)(u >> 16);
}
static __device__ __forceinline__ float bf16_to_f32(unsigned short h) {
    return __builtin_bit_cast(float, ((unsigned int)h) << 16);
}
static __device__ __forceinline__ v16bf pack_frag(u32x4 lo, u32x4 hi4) {
    return __builtin_bit_cast(
        v16bf, __builtin_shufflevector(lo, hi4, 0, 1, 2, 3, 4, 5, 6, 7));
}

// ---------------------------------------------------------------------------
// bf16 WMMA GEMM:  C(M,N) = A(M,K) * B(K,N) [+ bias] [leaky]
//   - one wave per (16*MT x 16) output block; B fragment loaded once per
//     K-chunk and reused for MT row-tiles (MT back-to-back WMMAs).
//   - A  : row-major bf16 (lda), contiguous per lane -> 2 x b128 per chunk
//   - Bt : TRANSPOSED weights (N x K) bf16, contiguous  -> 2 x b128 per chunk
// VGPR packing used here (contiguous k, same for A and B):
//   A 16x32 : lane half hi: elem e -> k = hi*16 + e  (lane = row)
//   B 32x16 : lane half hi: elem e -> k = hi*16 + e  (lane = column)
//   C/D     : vgpr j -> row m0 + hi*8 + j, col n0 + (lane&15)
// ---------------------------------------------------------------------------
template <int MT>
__global__ void gemm_bf16_wmma(const unsigned short* __restrict__ A, int lda,
                               const unsigned short* __restrict__ Bt,
                               const float* __restrict__ bias,
                               float* __restrict__ C, long ldc,
                               unsigned short* __restrict__ Cbf, long ldcbf,
                               int K, int act)
{
    const int lane = threadIdx.x;       // blockDim.x == 32 (one wave)
    const int lr   = lane & 15;
    const int hi   = lane >> 4;
    const int n0   = blockIdx.x * 16;
    const int m0   = blockIdx.y * (16 * MT);

    v8f acc[MT];
    const unsigned short* Arow[MT];
    #pragma unroll
    for (int mt = 0; mt < MT; ++mt) {
        acc[mt]  = v8f{};
        Arow[mt] = A + (long)(m0 + mt * 16 + lr) * lda;
    }
    const unsigned short* Brow = Bt + (long)(n0 + lr) * K;

    for (int k0 = 0; k0 < K; k0 += 32) {
        const int kb = k0 + hi * 16;                   // B: k = kb + e
        const u32x4 b_lo = *(const u32x4*)(Brow + kb);
        const u32x4 b_hi = *(const u32x4*)(Brow + kb + 8);
        const v16bf bfrag = pack_frag(b_lo, b_hi);

        const int ka = k0 + hi * 16;                   // A: k = ka + e
        #pragma unroll
        for (int mt = 0; mt < MT; ++mt) {
            const u32x4 a_lo = *(const u32x4*)(Arow[mt] + ka);
            const u32x4 a_hi = *(const u32x4*)(Arow[mt] + ka + 8);
            const v16bf afrag = pack_frag(a_lo, a_hi);
            acc[mt] = __builtin_amdgcn_wmma_f32_16x16x32_bf16(
                          false, afrag, false, bfrag, (short)0, acc[mt],
                          false, false);
        }
    }

    const int col = n0 + lr;
    const float bv = bias ? bias[col] : 0.0f;
    #pragma unroll
    for (int mt = 0; mt < MT; ++mt) {
        #pragma unroll
        for (int j = 0; j < 8; ++j) {
            const int row = m0 + mt * 16 + hi * 8 + j;
            float v = acc[mt][j] + bv;
            if (act == 1) v = (v > 0.0f) ? v : 0.1f * v;   // LeakyReLU(0.1)
            if (C)   C[(long)row * ldc + col]     = v;
            if (Cbf) Cbf[(long)row * ldcbf + col] = f32_to_bf16(v);
        }
    }
}

// ---------------------------------------------------------------------------
// f32 -> raw bf16 conversion (row-major copy)
// ---------------------------------------------------------------------------
__global__ void cvt_f32_bf16(const float* __restrict__ src,
                             unsigned short* __restrict__ dst, long n)
{
    long i = (long)blockIdx.x * blockDim.x + threadIdx.x;
    if (i < n) dst[i] = f32_to_bf16(src[i]);
}

// f32 (K,N) -> bf16 transposed (N,K).  One-time cost; coalesced reads.
__global__ void cvt_t_f32_bf16(const float* __restrict__ src,
                               unsigned short* __restrict__ dst,
                               int Kd, int Nd)
{
    long i = (long)blockIdx.x * blockDim.x + threadIdx.x;
    const long n = (long)Kd * Nd;
    if (i < n) {
        const int k = (int)(i / Nd);
        const int c = (int)(i % Nd);
        dst[(long)c * Kd + k] = f32_to_bf16(src[i]);
    }
}

// ---------------------------------------------------------------------------
// Initialize recurrent state: h (f32 + bf16) and out_prev (bf16 half of x).
// grid = 32 (batch), block = 512
// ---------------------------------------------------------------------------
__global__ void init_state(const float* __restrict__ h0,
                           const float* __restrict__ out0,
                           float* __restrict__ h,
                           unsigned short* __restrict__ hbf,
                           unsigned short* __restrict__ xbuf)
{
    const int b = blockIdx.x, j = threadIdx.x;
    xbuf[b * XIN + E_DIM + j] = f32_to_bf16(out0[b * OUT_D_N + j]);
    if (j < UNITS_N) {
        const float v = h0[b * UNITS_N + j];
        h[b * UNITS_N + j]   = v;
        hbf[b * UNITS_N + j] = f32_to_bf16(v);
    }
}

// ---------------------------------------------------------------------------
// Attention for one step: scores -> softmax -> es output -> context c.
// grid = 32 (batch), block = 256.  WaS/enc are bf16 and L2-resident.
// ---------------------------------------------------------------------------
__global__ void attn_kernel(const unsigned short* __restrict__ WaS,   // (B,Tenc,E)
                            const unsigned short* __restrict__ encbf, // (B,Tenc,E)
                            const float* __restrict__ hU,             // (B,E)
                            const float* __restrict__ Va,             // (E)
                            float* __restrict__ es_out,               // (B,Tdec,Tenc)
                            unsigned short* __restrict__ xbuf,        // (B,XIN)
                            unsigned short* __restrict__ cat1,        // (B,CAT1)
                            int t_step)
{
    const int b = blockIdx.x, tid = threadIdx.x;
    __shared__ float hU_s[E_DIM];
    __shared__ float Va_s[E_DIM];
    __shared__ float e_s[T_ENC];
    __shared__ float red[256];

    for (int i = tid; i < E_DIM; i += 256) {
        hU_s[i] = hU[b * E_DIM + i];
        Va_s[i] = Va[i];
    }
    __syncthreads();

    // --- scores: tanh(WaS + hU) . Va over E; 4 rows per thread, b128 loads
    float sc[4];
    const unsigned short* wb = WaS + (long)b * T_ENC * E_DIM;
    #pragma unroll
    for (int i = 0; i < 4; ++i) {
        const int t = tid + i * 256;
        const unsigned short* wr = wb + (long)t * E_DIM;
        float acc = 0.0f;
        for (int e = 0; e < E_DIM; e += 8) {
            const u32x4 u = *(const u32x4*)(wr + e);
            #pragma unroll
            for (int j = 0; j < 4; ++j) {
                const float w0 = __builtin_bit_cast(float, u[j] << 16);
                const float w1 = __builtin_bit_cast(float, u[j] & 0xffff0000u);
                acc += tanhf(w0 + hU_s[e + 2 * j])     * Va_s[e + 2 * j];
                acc += tanhf(w1 + hU_s[e + 2 * j + 1]) * Va_s[e + 2 * j + 1];
            }
        }
        sc[i] = acc;
    }

    // --- softmax over T_ENC ----------------------------------------------
    float lm = fmaxf(fmaxf(sc[0], sc[1]), fmaxf(sc[2], sc[3]));
    red[tid] = lm;  __syncthreads();
    for (int s = 128; s > 0; s >>= 1) {
        if (tid < s) red[tid] = fmaxf(red[tid], red[tid + s]);
        __syncthreads();
    }
    const float m = red[0];
    __syncthreads();

    float ls = 0.0f;
    #pragma unroll
    for (int i = 0; i < 4; ++i) {
        const float ex = __expf(sc[i] - m);
        e_s[tid + i * 256] = ex;
        ls += ex;
    }
    red[tid] = ls;  __syncthreads();
    for (int s = 128; s > 0; s >>= 1) {
        if (tid < s) red[tid] += red[tid + s];
        __syncthreads();
    }
    const float inv = 1.0f / red[0];

    float* eo = es_out + ((long)b * T_DEC + t_step) * T_ENC;
    #pragma unroll
    for (int i = 0; i < 4; ++i) {
        const int t = tid + i * 256;
        const float en = e_s[t] * inv;
        e_s[t] = en;
        eo[t]  = en;
    }
    __syncthreads();

    // --- context c = e . enc : 2 adjacent cols/thread, coalesced b32 rows
    const unsigned short* eb = encbf + (long)b * T_ENC * E_DIM;
    const int c0 = 2 * tid;
    float a0 = 0.0f, a1 = 0.0f;
    for (int t = 0; t < T_ENC; ++t) {
        const unsigned int u = *(const unsigned int*)(eb + (long)t * E_DIM + c0);
        const float w = e_s[t];
        a0 += w * __builtin_bit_cast(float, u << 16);
        a1 += w * __builtin_bit_cast(float, u & 0xffff0000u);
    }
    const unsigned short cb0 = f32_to_bf16(a0);
    const unsigned short cb1 = f32_to_bf16(a1);
    xbuf[b * XIN + c0]     = cb0;                          // GRU input half
    xbuf[b * XIN + c0 + 1] = cb1;
    cat1[b * CAT1 + D_DIM + UNITS_N + c0]     = cb0;       // dense1 last third
    cat1[b * CAT1 + D_DIM + UNITS_N + c0 + 1] = cb1;
}

// ---------------------------------------------------------------------------
// GRU gates (reset_after=True) + assemble cat1 = [x_t, h_new, c].
// grid = 32 (batch), block = 256 (units)
// ---------------------------------------------------------------------------
__global__ void gates_kernel(const float* __restrict__ xz,   // (B,3U) incl bias0
                             const float* __restrict__ hz,   // (B,3U) incl bias1
                             float* __restrict__ h,          // (B,U) in/out
                             unsigned short* __restrict__ hbf,
                             unsigned short* __restrict__ cat1,
                             const unsigned short* __restrict__ decbf, // (B,Tdec,D)
                             int t_step)
{
    const int b = blockIdx.x, u = threadIdx.x;
    const long o = (long)b * GRU3;
    const float z  = 1.0f / (1.0f + __expf(-(xz[o + u]           + hz[o + u])));
    const float r  = 1.0f / (1.0f + __expf(-(xz[o + UNITS_N + u] + hz[o + UNITS_N + u])));
    const float hh = tanhf(xz[o + 2 * UNITS_N + u] + r * hz[o + 2 * UNITS_N + u]);
    const float hp = h[b * UNITS_N + u];
    const float hn = z * hp + (1.0f - z) * hh;
    h[b * UNITS_N + u] = hn;
    const unsigned short hb = f32_to_bf16(hn);
    hbf[b * UNITS_N + u]       = hb;
    cat1[b * CAT1 + D_DIM + u] = hb;                                    // h_new
    cat1[b * CAT1 + u] = decbf[((long)b * T_DEC + t_step) * D_DIM + u]; // x_t
}

// ---------------------------------------------------------------------------
// Host-side orchestration
// ---------------------------------------------------------------------------
extern "C" void kernel_launch(void* const* d_in, const int* in_sizes, int n_in,
                              void* d_out, int out_size, void* d_ws, size_t ws_size,
                              hipStream_t stream)
{
    const float* enc   = (const float*)d_in[0];   // (32,1024,512)
    const float* dec   = (const float*)d_in[1];   // (32,256,256)
    const float* h0    = (const float*)d_in[2];   // (32,256)
    const float* out0  = (const float*)d_in[4];   // (32,512)
    const float* W_a   = (const float*)d_in[5];   // (512,512)
    const float* U_a   = (const float*)d_in[6];   // (256,512)
    const float* V_a   = (const float*)d_in[7];   // (512,1)
    const float* gruK  = (const float*)d_in[8];   // (1024,768)
    const float* gruR  = (const float*)d_in[9];   // (256,768)
    const float* gbias = (const float*)d_in[10];  // (2,768)
    const float* W1    = (const float*)d_in[11];  // (1024,512)
    const float* b1    = (const float*)d_in[12];  // (512)
    const float* W2    = (const float*)d_in[13];  // (512,512)
    const float* b2    = (const float*)d_in[14];  // (512)

    float* outs = (float*)d_out;                              // (32,256,512)
    float* es   = outs + (long)B_SZ * T_DEC * OUT_D_N;        // (32,256,1024)

    // ---- workspace layout (256-byte aligned slabs) ----------------------
    char* ws = (char*)d_ws;
    size_t off = 0;
    auto alloc = [&](size_t bytes) -> char* {
        char* p = ws + off;
        off += (bytes + 255) & ~(size_t)255;
        return p;
    };
    unsigned short* enc_bf = (unsigned short*)alloc((size_t)B_SZ * T_ENC * E_DIM * 2);
    unsigned short* WaS_bf = (unsigned short*)alloc((size_t)B_SZ * T_ENC * E_DIM * 2);
    unsigned short* dec_bf = (unsigned short*)alloc((size_t)B_SZ * T_DEC * D_DIM * 2);
    unsigned short* WaT_bf = (unsigned short*)alloc((size_t)E_DIM * E_DIM * 2);      // (N=512,K=512)
    unsigned short* UaT_bf = (unsigned short*)alloc((size_t)E_DIM * UNITS_N * 2);    // (N=512,K=256)
    unsigned short* gKT_bf = (unsigned short*)alloc((size_t)GRU3 * XIN * 2);         // (N=768,K=1024)
    unsigned short* gRT_bf = (unsigned short*)alloc((size_t)GRU3 * UNITS_N * 2);     // (N=768,K=256)
    unsigned short* W1T_bf = (unsigned short*)alloc((size_t)OUT_D_N * CAT1 * 2);     // (N=512,K=1024)
    unsigned short* W2T_bf = (unsigned short*)alloc((size_t)OUT_D_N * OUT_D_N * 2);  // (N=512,K=512)
    float*          h_f    = (float*)alloc((size_t)B_SZ * UNITS_N * 4);
    unsigned short* h_bf   = (unsigned short*)alloc((size_t)B_SZ * UNITS_N * 2);
    unsigned short* x_bf   = (unsigned short*)alloc((size_t)B_SZ * XIN * 2);
    unsigned short* c1_bf  = (unsigned short*)alloc((size_t)B_SZ * CAT1 * 2);
    float*          hU_f   = (float*)alloc((size_t)B_SZ * E_DIM * 4);
    float*          xz_f   = (float*)alloc((size_t)B_SZ * GRU3 * 4);
    float*          hz_f   = (float*)alloc((size_t)B_SZ * GRU3 * 4);
    unsigned short* act_bf = (unsigned short*)alloc((size_t)B_SZ * OUT_D_N * 2);
    (void)ws_size; (void)n_in; (void)in_sizes; (void)out_size;

    // ---- one-time conversions -------------------------------------------
    {   // row-major bf16 copies (A-side / attention data)
        const long n_enc = (long)B_SZ * T_ENC * E_DIM;
        const long n_dec = (long)B_SZ * T_DEC * D_DIM;
        cvt_f32_bf16<<<(int)((n_enc + 255) / 256), 256, 0, stream>>>(enc, enc_bf, n_enc);
        cvt_f32_bf16<<<(int)((n_dec + 255) / 256), 256, 0, stream>>>(dec, dec_bf, n_dec);
    }
    struct TJob { const float* s; unsigned short* d; int K, N; };
    const TJob tjobs[] = {
        { W_a,  WaT_bf, E_DIM,   E_DIM },
        { U_a,  UaT_bf, UNITS_N, E_DIM },
        { gruK, gKT_bf, XIN,     GRU3 },
        { gruR, gRT_bf, UNITS_N, GRU3 },
        { W1,   W1T_bf, CAT1,    OUT_D_N },
        { W2,   W2T_bf, OUT_D_N, OUT_D_N },
    };
    for (const auto& j : tjobs) {
        const long n = (long)j.K * j.N;
        cvt_t_f32_bf16<<<(int)((n + 255) / 256), 256, 0, stream>>>(j.s, j.d, j.K, j.N);
    }
    init_state<<<B_SZ, 512, 0, stream>>>(h0, out0, h_f, h_bf, x_bf);

    // ---- WaS = enc @ W_a  (32768 x 512, K=512): 8 row-tiles per wave ----
    gemm_bf16_wmma<8><<<dim3(E_DIM / 16, (B_SZ * T_ENC) / 128), 32, 0, stream>>>(
        enc_bf, E_DIM, WaT_bf, nullptr,
        nullptr, 0, WaS_bf, E_DIM, E_DIM, 0);

    // ---- 256 sequential decode steps ------------------------------------
    for (int t = 0; t < T_DEC; ++t) {
        // hU = h @ U_a    (32 x 512, K=256)
        gemm_bf16_wmma<2><<<dim3(E_DIM / 16, 1), 32, 0, stream>>>(
            h_bf, UNITS_N, UaT_bf, nullptr,
            hU_f, E_DIM, nullptr, 0, UNITS_N, 0);

        // attention: scores -> softmax (es output) -> context c
        attn_kernel<<<B_SZ, 256, 0, stream>>>(
            WaS_bf, enc_bf, hU_f, V_a, es, x_bf, c1_bf, t);

        // xz = [c, out_prev] @ gru_kernel + bias0   (32 x 768, K=1024)
        gemm_bf16_wmma<2><<<dim3(GRU3 / 16, 1), 32, 0, stream>>>(
            x_bf, XIN, gKT_bf, gbias,
            xz_f, GRU3, nullptr, 0, XIN, 0);

        // hz = h @ gru_rec_kernel + bias1           (32 x 768, K=256)
        gemm_bf16_wmma<2><<<dim3(GRU3 / 16, 1), 32, 0, stream>>>(
            h_bf, UNITS_N, gRT_bf, gbias + GRU3,
            hz_f, GRU3, nullptr, 0, UNITS_N, 0);

        // gates -> h_new; assemble cat1 = [x_t, h_new, c]
        gates_kernel<<<B_SZ, UNITS_N, 0, stream>>>(
            xz_f, hz_f, h_f, h_bf, c1_bf, dec_bf, t);

        // dense1 + LeakyReLU -> bf16 activations   (32 x 512, K=1024)
        gemm_bf16_wmma<2><<<dim3(OUT_D_N / 16, 1), 32, 0, stream>>>(
            c1_bf, CAT1, W1T_bf, b1,
            nullptr, 0, act_bf, OUT_D_N, CAT1, 1);

        // dense2 -> outs[:, t, :] (f32) and out_prev (bf16 into x 2nd half)
        gemm_bf16_wmma<2><<<dim3(OUT_D_N / 16, 1), 32, 0, stream>>>(
            act_bf, OUT_D_N, W2T_bf, b2,
            outs + (long)t * OUT_D_N, (long)T_DEC * OUT_D_N,
            x_bf + E_DIM, XIN, OUT_D_N, 0);
    }
}
